// LMPNN_30485677867316
// MI455X (gfx1250) — matrix operand order repaired and checked by reading, compiled, not run on hardware
//
#include <hip/hip_runtime.h>

typedef __bf16 bf16;
typedef __attribute__((ext_vector_type(16))) __bf16 v16bf;
typedef __attribute__((ext_vector_type(8)))  __bf16 v8bf;
typedef __attribute__((ext_vector_type(8)))  float  v8f;

#define N_NODES 4096
#define N_EDGES 262144
#define DIM     128
#define N_ENT   50000

#define MT      64                   // node (N-dim) tile per workgroup
#define ET      64                   // entity tile
#define ESPLIT  8                    // entity-range splits across blockIdx.y
#define TPS     98                   // entity tiles per split
#define ENT_PAD (ESPLIT * TPS * ET)  // 50176 padded entity rows (zeros beyond N_ENT)

// workspace layout (bytes)
#define OFF_AGGR 0                                         // 4096*128 f32 = 2 MB
#define OFF_ENTB (OFF_AGGR + (size_t)N_NODES * DIM * 4)    // ENT_PAD*128 bf16
#define OFF_DEG  (OFF_ENTB + (size_t)ENT_PAD * DIM * 2)    // 4096 i32
#define OFF_OFF  (OFF_DEG + (size_t)N_NODES * 4)           // 4097 i32
#define OFF_CUR  (OFF_OFF + (size_t)(N_NODES + 4) * 4)     // 4096 i32
#define OFF_EL   (OFF_CUR + (size_t)N_NODES * 4)           // 262144 i32

// ---------------------------------------------------------------------------
// Kernel 1: out = 0 ; deg = 0
// ---------------------------------------------------------------------------
__global__ __launch_bounds__(256) void k_init(float* __restrict__ out,
                                              int* __restrict__ deg) {
  int i = blockIdx.x * 256 + threadIdx.x;
  out[i] = 0.0f;
  if (i < N_NODES) deg[i] = 0;
}

// ---------------------------------------------------------------------------
// Kernel 2: count in-degree per destination node
// ---------------------------------------------------------------------------
__global__ __launch_bounds__(256) void k_count(const int* __restrict__ ei,
                                               int* __restrict__ deg) {
  int e = blockIdx.x * 256 + threadIdx.x;
  atomicAdd(&deg[ei[N_EDGES + e]], 1);
}

// ---------------------------------------------------------------------------
// Kernel 3: single-block exclusive scan of deg -> off, cursor
// ---------------------------------------------------------------------------
__global__ __launch_bounds__(256) void k_scan(const int* __restrict__ deg,
                                              int* __restrict__ off,
                                              int* __restrict__ cursor) {
  __shared__ int vals[N_NODES];
  __shared__ int part[256];
  const int t = threadIdx.x;
  const int base = t * 16;
  int s = 0;
#pragma unroll
  for (int i = 0; i < 16; ++i) { int v = deg[base + i]; vals[base + i] = v; s += v; }
  part[t] = s;
  __syncthreads();
  if (t == 0) {
    int run = 0;
    for (int i = 0; i < 256; ++i) { int v = part[i]; part[i] = run; run += v; }
  }
  __syncthreads();
  int run = part[t];
#pragma unroll
  for (int i = 0; i < 16; ++i) {
    int v = vals[base + i];
    off[base + i] = run;
    cursor[base + i] = run;
    run += v;
  }
  if (t == 255) off[N_NODES] = run;
}

// ---------------------------------------------------------------------------
// Kernel 4: scatter edge ids into dst-major CSR lists
// ---------------------------------------------------------------------------
__global__ __launch_bounds__(256) void k_scatter(const int* __restrict__ ei,
                                                 int* __restrict__ cursor,
                                                 int* __restrict__ elist) {
  int e = blockIdx.x * 256 + threadIdx.x;
  int slot = atomicAdd(&cursor[ei[N_EDGES + e]], 1);
  elist[slot] = e;
}

// ---------------------------------------------------------------------------
// Kernel 5: gather aggregation, one wave per node, zero atomics.
// aggr[n] = 0.1*x[n] + sum_{e: dst(e)=n} (x[src(e)] + rel_emb[rel(e)]) * coef(e)
// ---------------------------------------------------------------------------
__global__ __launch_bounds__(256) void k_gather(const float* __restrict__ x,
                                                const int* __restrict__ ei,
                                                const int* __restrict__ rel,
                                                const int* __restrict__ neg,
                                                const float* __restrict__ rel_emb,
                                                const int* __restrict__ off,
                                                const int* __restrict__ elist,
                                                float* __restrict__ aggr) {
  const int lane = threadIdx.x & 31;
  const int w    = threadIdx.x >> 5;
  const int n    = blockIdx.x * 8 + w;
  const int d    = lane * 4;

  const float4 x0 = *(const float4*)(x + (size_t)n * DIM + d);
  float a0 = 0.1f * x0.x, a1 = 0.1f * x0.y, a2 = 0.1f * x0.z, a3 = 0.1f * x0.w;

  const int beg = off[n], end = off[n + 1];
  for (int i = beg; i < end; ++i) {
    const int e = elist[i];
    const int src = ei[e];
    const int r = rel[e];
    const float coef = 1.0f - 2.0f * (float)neg[e];
    const float4 xv = *(const float4*)(x + (size_t)src * DIM + d);
    const float4 rv = *(const float4*)(rel_emb + (size_t)r * DIM + d);
    a0 += (xv.x + rv.x) * coef;
    a1 += (xv.y + rv.y) * coef;
    a2 += (xv.z + rv.z) * coef;
    a3 += (xv.w + rv.w) * coef;
  }
  float4 o; o.x = a0; o.y = a1; o.z = a2; o.w = a3;
  *(float4*)(aggr + (size_t)n * DIM + d) = o;
}

// ---------------------------------------------------------------------------
// Kernel 6: entity_emb f32 -> bf16, zero-padded to ENT_PAD rows.
// ---------------------------------------------------------------------------
__global__ __launch_bounds__(256) void k_entcvt(const float* __restrict__ ent,
                                                bf16* __restrict__ entb) {
  unsigned tid = blockIdx.x * 256u + threadIdx.x;
  unsigned i4 = tid * 4u;
  unsigned n = i4 / DIM;
  unsigned d = i4 % DIM;
  bf16* o = entb + i4;
  if (n < N_ENT) {
    const float4 v = *(const float4*)(ent + (size_t)n * DIM + d);
    o[0] = (bf16)v.x; o[1] = (bf16)v.y; o[2] = (bf16)v.z; o[3] = (bf16)v.w;
  } else {
    o[0] = (bf16)0.0f; o[1] = (bf16)0.0f; o[2] = (bf16)0.0f; o[3] = (bf16)0.0f;
  }
}

// ---------------------------------------------------------------------------
// Kernel 7: fused  S^T = E @ aggr^T ; S' = relu(S*scale+bias) ; O^T += E^T @ S'^T
// Both GEMMs transposed so that A/B fragments AND the epilogue store are
// contiguous in LDS (epilogue = one packed ds_store_b128 per tile).
// Wave w: g = w>>1 (entity strip in GEMM1, node column in GEMM2), half = w&1.
// ---------------------------------------------------------------------------
__global__ __launch_bounds__(256)
void k_fused(const float* __restrict__ aggr,
             const bf16* __restrict__ entb,
             const float* __restrict__ scale,
             const float* __restrict__ bias,
             float* __restrict__ out) {
  __shared__ bf16 En[2][ET][DIM];   // entity tile, row-major   (2 x 16 KB)
  __shared__ bf16 Et[2][DIM][ET];   // entity tile, transposed  (2 x 16 KB)
  __shared__ bf16 Ssh[MT][ET];      // relu'd scores, node-major (8 KB)

  const int t    = threadIdx.x;
  const int lane = t & 31;
  const int w    = t >> 5;
  const int ln   = lane & 15;    // fragment row/col within 16
  const int h    = lane >> 4;    // lane half
  const int g    = w >> 1;       // 0..3
  const int half = w & 1;

  const int m_base = blockIdx.x * MT;

  // staging sub-block: 4 entities x 8 dims per thread
  const int e0  = (t & 15) * 4;
  const int d0s = (t >> 4) * 8;

  // ---- preload B1 fragments (f32 aggr -> bf16) ----
  // B fragment (contiguous K order, same as A): lane ln = column (node), elems j -> K = kc*32 + h*16 + j
  // wave covers node columns half*32 + {0,16} + ln
  v16bf b1[2][4];
#pragma unroll
  for (int c = 0; c < 2; ++c) {
    const float* arow = aggr + (size_t)(m_base + half * 32 + c * 16 + ln) * DIM;
#pragma unroll
    for (int kc = 0; kc < 4; ++kc) {
      const float* pp = arow + kc * 32 + h * 16;
      v16bf b;
#pragma unroll
      for (int j = 0; j < 16; ++j) b[j] = (bf16)pp[j];
      b1[c][kc] = b;
    }
  }

  // O^T accumulators: 4 tiles, rows d = half*64 + tt*16 + (h*8+r), col node g*16+ln
  v8f oacc[4];
#pragma unroll
  for (int i = 0; i < 4; ++i)
#pragma unroll
    for (int r = 0; r < 8; ++r) oacc[i][r] = 0.0f;

  const int tile0 = blockIdx.y * TPS;

  // ---- prologue: load tile 0 into registers ----
  union Q { uint4 u; unsigned short s[8]; };
  Q q[4];
  {
    const bf16* gp = entb + (size_t)(tile0 * ET + e0) * DIM + d0s;
    q[0].u = *(const uint4*)(gp);
    q[1].u = *(const uint4*)(gp + DIM);
    q[2].u = *(const uint4*)(gp + 2 * DIM);
    q[3].u = *(const uint4*)(gp + 3 * DIM);
  }

  for (int it = 0; it < TPS; ++it) {
    const int p = it & 1;
    const int ebase = (tile0 + it) * ET;

    // ---- write staged registers into LDS buffers[p] ----
#pragma unroll
    for (int i = 0; i < 4; ++i) *(uint4*)&En[p][e0 + i][d0s] = q[i].u;
#pragma unroll
    for (int d = 0; d < 8; ++d) {
      uint2 pk;
      pk.x = (unsigned)q[0].s[d] | ((unsigned)q[1].s[d] << 16);
      pk.y = (unsigned)q[2].s[d] | ((unsigned)q[3].s[d] << 16);
      *(uint2*)&Et[p][d0s + d][e0] = pk;
    }

    // ---- issue next tile's global loads (latency hides behind WMMAs) ----
    if (it + 1 < TPS) {
      const bf16* gp = entb + (size_t)(ebase + ET + e0) * DIM + d0s;
      q[0].u = *(const uint4*)(gp);
      q[1].u = *(const uint4*)(gp + DIM);
      q[2].u = *(const uint4*)(gp + 2 * DIM);
      q[3].u = *(const uint4*)(gp + 3 * DIM);
      if (it + 2 < TPS) __builtin_prefetch(gp + (size_t)ET * DIM, 0, 0);
    }
    __syncthreads();   // barrier A: buffers[p] visible; prev-iter Ssh reads done

    // ---- GEMM1: S^T = En x b1   (entity strip g, node cols half*32+{0,16}) ----
    v8f s0, s1;
#pragma unroll
    for (int r = 0; r < 8; ++r) { s0[r] = 0.0f; s1[r] = 0.0f; }
#pragma unroll
    for (int kc = 0; kc < 4; ++kc) {
      // A fragment: lane ln = entity row, K = kc*32 + h*16 + j (contiguous)
      const bf16* arow = &En[p][g * 16 + ln][kc * 32];
      v8bf lo = *(const v8bf*)(arow + h * 16);
      v8bf hi = *(const v8bf*)(arow + 8 + h * 16);
      v16bf a1;
#pragma unroll
      for (int j = 0; j < 8; ++j) { a1[j] = lo[j]; a1[j + 8] = hi[j]; }
      s0 = __builtin_amdgcn_wmma_f32_16x16x32_bf16(false, a1, false, b1[0][kc],
                                                   (short)0, s0, false, false);
      s1 = __builtin_amdgcn_wmma_f32_16x16x32_bf16(false, a1, false, b1[1][kc],
                                                   (short)0, s1, false, false);
    }

    // ---- epilogue: relu(s*scale+bias) -> packed bf16 row store into Ssh ----
    {
      // this wave's 8 entity rows: ebase + g*16 + h*8 + r  (8-aligned block;
      // N_ENT % 8 == 0 so a block is entirely in- or out-of-range)
      const int eb = ebase + g * 16 + h * 8;
      const float* sp = scale + (eb < N_ENT ? eb : 0);
      const float* bp = bias + (eb < N_ENT ? eb : 0);
      const float4 sA = *(const float4*)(sp), sB = *(const float4*)(sp + 4);
      const float4 bA = *(const float4*)(bp), bB = *(const float4*)(bp + 4);
      const float scv[8] = {sA.x, sA.y, sA.z, sA.w, sB.x, sB.y, sB.z, sB.w};
      const float biv[8] = {bA.x, bA.y, bA.z, bA.w, bB.x, bB.y, bB.z, bB.w};
      v8bf o0, o1;
#pragma unroll
      for (int r = 0; r < 8; ++r) {
        float v0 = s0[r] * scv[r] + biv[r]; v0 = v0 > 0.0f ? v0 : 0.0f;
        float v1 = s1[r] * scv[r] + biv[r]; v1 = v1 > 0.0f ? v1 : 0.0f;
        o0[r] = (bf16)v0;
        o1[r] = (bf16)v1;
      }
      *(v8bf*)&Ssh[half * 32 + ln][g * 16 + h * 8]      = o0;
      *(v8bf*)&Ssh[half * 32 + 16 + ln][g * 16 + h * 8] = o1;
    }
    __syncthreads();   // barrier B: Ssh visible

    // ---- GEMM2: O^T += Et x Ssh  (d rows half*64+tt*16, node col g) ----
#pragma unroll
    for (int kc = 0; kc < 2; ++kc) {
      // B fragment: lane ln = node col, K = kc*32 + h*16 + j (contiguous)
      v16bf b2 = *(const v16bf*)&Ssh[g * 16 + ln][kc * 32 + h * 16];
#pragma unroll
      for (int tt = 0; tt < 4; ++tt) {
        // A fragment: lane ln = d row, K = kc*32 + h*16 + j (contiguous)
        const bf16* erow = &Et[p][half * 64 + tt * 16 + ln][kc * 32];
        v8bf lo = *(const v8bf*)(erow + h * 16);
        v8bf hi = *(const v8bf*)(erow + 8 + h * 16);
        v16bf a2;
#pragma unroll
        for (int j = 0; j < 8; ++j) { a2[j] = lo[j]; a2[j + 8] = hi[j]; }
        oacc[tt] = __builtin_amdgcn_wmma_f32_16x16x32_bf16(false, a2, false, b2,
                                                           (short)0, oacc[tt],
                                                           false, false);
      }
    }
    // no barrier: next iteration writes the other ping-pong buffer
  }

  // ---- accumulate this entity-split's contribution into out ----
  {
    const int m = m_base + g * 16 + ln;
#pragma unroll
    for (int tt = 0; tt < 4; ++tt) {
#pragma unroll
      for (int r = 0; r < 8; ++r) {
        const int d = half * 64 + tt * 16 + h * 8 + r;
        atomicAdd(out + (size_t)m * DIM + d, oacc[tt][r]);
      }
    }
  }
}

// ---------------------------------------------------------------------------
extern "C" void kernel_launch(void* const* d_in, const int* in_sizes, int n_in,
                              void* d_out, int out_size, void* d_ws, size_t ws_size,
                              hipStream_t stream) {
  (void)in_sizes; (void)n_in; (void)out_size; (void)ws_size;
  const float* x           = (const float*)d_in[0];
  const int*   edge_index  = (const int*)d_in[1];
  const int*   relation_id = (const int*)d_in[2];
  const int*   neg_flag    = (const int*)d_in[3];
  const float* rel_emb     = (const float*)d_in[4];
  const float* entity_emb  = (const float*)d_in[5];
  const float* scale       = (const float*)d_in[6];
  const float* bias        = (const float*)d_in[7];
  float* out = (float*)d_out;

  char* ws = (char*)d_ws;
  float* aggr  = (float*)(ws + OFF_AGGR);
  bf16*  entb  = (bf16*)(ws + OFF_ENTB);
  int*   deg   = (int*)(ws + OFF_DEG);
  int*   off   = (int*)(ws + OFF_OFF);
  int*   cur   = (int*)(ws + OFF_CUR);
  int*   elist = (int*)(ws + OFF_EL);

  k_init   <<<(N_NODES * DIM) / 256, 256, 0, stream>>>(out, deg);
  k_count  <<<N_EDGES / 256, 256, 0, stream>>>(edge_index, deg);
  k_scan   <<<1, 256, 0, stream>>>(deg, off, cur);
  k_scatter<<<N_EDGES / 256, 256, 0, stream>>>(edge_index, cur, elist);
  k_gather <<<N_NODES / 8, 256, 0, stream>>>(x, edge_index, relation_id,
                                             neg_flag, rel_emb, off, elist, aggr);
  k_entcvt <<<((size_t)ENT_PAD * DIM / 4) / 256, 256, 0, stream>>>(entity_emb, entb);
  k_fused  <<<dim3(N_NODES / MT, ESPLIT), 256, 0, stream>>>(aggr, entb, scale,
                                                            bias, out);
}
